// MoELayer_16466904613124
// MI455X (gfx1250) — matrix operand order repaired and checked by reading, not compiled, not run on hardware
//
#include <hip/hip_runtime.h>
#include <math.h>

// Problem constants (B=4, S=2048 -> T=8192)
#define TOK   8192
#define DH    1024      // H
#define DI    2048      // I
#define NE    8         // experts
#define TOPK  2
#define MT    32        // tokens per FFN block (2 wmma m-tiles)

typedef __attribute__((ext_vector_type(16))) __bf16 v16bf;
typedef __attribute__((ext_vector_type(8)))  float  v8f;

// ---- d_out layout (float offsets): output | expert_frac | avg_prob | logits | probs
#define OFF_OUT   0
#define OFF_EF    (TOK*DH)
#define OFF_AP    (OFF_EF + NE)
#define OFF_LOG   (OFF_AP + NE)
#define OFF_PROB  (OFF_LOG + TOK*NE)

// ---- workspace layout (byte offsets)
#define WS_COUNTS   0                         // int[NE]
#define WS_OFFSETS  64                        // int[NE]
#define WS_TOPE     128                       // int[TOK*TOPK]
#define WS_TOPW     (WS_TOPE + TOK*TOPK*4)    // float[TOK*TOPK]
#define WS_PTOK     (WS_TOPW + TOK*TOPK*4)    // int[TOK*TOPK]
#define WS_PW       (WS_PTOK + TOK*TOPK*4)    // float[TOK*TOPK]

// LDS row strides (bf16 elements), padded to dodge bank conflicts
#define XS (DH + 8)
#define HS (DI + 8)

// ---------------- WMMA fragment helpers (CDNA5 16x16x32 bf16 layouts) -------
// A-matrix 16x32 (ISA 7.12.2): lane<16 -> M=lane, K in {0..7,16..23};
// lane>=16 -> M=lane-16, K in {8..15,24..31}; 2 K-values per VGPR.
__device__ __forceinline__ int fragA_k(int j, int half) {
    return ((j < 4) ? (2 * j) : (16 + 2 * (j - 4))) + half * 8;
}
// B-matrix 32x16: lane<16 -> N=lane, K in {0..7,16..23};
// lane>=16 -> N=lane-16, K in {8..15,24..31}; same K order as A.
__device__ __forceinline__ int fragB_k(int j, int half) {
    return ((j < 4) ? (2 * j) : (16 + 2 * (j - 4))) + half * 8;
}

__device__ __forceinline__ v16bf fragA_lds(const __bf16* row, int k0, int half) {
    v16bf f;
#pragma unroll
    for (int j = 0; j < 8; ++j) {
        int k = k0 + fragA_k(j, half);
        f[2 * j]     = row[k];
        f[2 * j + 1] = row[k + 1];
    }
    return f;
}

__device__ __forceinline__ v16bf fragB_g32(const float* __restrict__ row, int k0, int half) {
    v16bf f;
#pragma unroll
    for (int j = 0; j < 8; ++j) {
        int k = k0 + fragB_k(j, half);
        f[2 * j]     = (__bf16)row[k];
        f[2 * j + 1] = (__bf16)row[k + 1];
    }
    return f;
}

__device__ __forceinline__ v8f wmma_bf16(v16bf a, v16bf b, v8f c) {
    return __builtin_amdgcn_wmma_f32_16x16x32_bf16(false, a, false, b, (short)0, c,
                                                   false, false);
}

__device__ __forceinline__ float silu_mul(float g, float u) {
    // g * sigmoid(g) * u, with fast hardware reciprocal (v_rcp_f32)
    return g * __builtin_amdgcn_rcpf(1.0f + __expf(-g)) * u;
}

// ---------------- kernel 1: zero output region + expert counters ------------
__global__ void k_zero(float* __restrict__ out, int* __restrict__ counts) {
    int gid = blockIdx.x * blockDim.x + threadIdx.x;
    int stride = gridDim.x * blockDim.x;
    for (int i = gid; i < TOK * DH; i += stride) out[i] = 0.0f;
    if (gid < NE) counts[gid] = 0;
}

// ---------------- kernel 2: router (one wave32 per token) -------------------
__global__ void __launch_bounds__(256) k_router(
    const float* __restrict__ x, const float* __restrict__ rw,
    float* __restrict__ dout, int* __restrict__ counts,
    int* __restrict__ top_e, float* __restrict__ top_w) {
    const int lane  = threadIdx.x & 31;
    const int wv    = threadIdx.x >> 5;
    const int token = blockIdx.x * (blockDim.x >> 5) + wv;
    if (token >= TOK) return;

    float acc[NE];
#pragma unroll
    for (int e = 0; e < NE; ++e) acc[e] = 0.0f;
    const float* xr = x + (size_t)token * DH;
    for (int h = lane; h < DH; h += 32) {
        float xv = xr[h];
#pragma unroll
        for (int e = 0; e < NE; ++e) acc[e] += xv * rw[e * DH + h];
    }
#pragma unroll
    for (int e = 0; e < NE; ++e)
        for (int off = 16; off > 0; off >>= 1)
            acc[e] += __shfl_xor(acc[e], off, 32);

    if (lane == 0) {
        float mx = acc[0];
#pragma unroll
        for (int e = 1; e < NE; ++e) mx = fmaxf(mx, acc[e]);
        float p[NE], s = 0.0f;
#pragma unroll
        for (int e = 0; e < NE; ++e) { p[e] = __expf(acc[e] - mx); s += p[e]; }
        float inv = 1.0f / s;
#pragma unroll
        for (int e = 0; e < NE; ++e) {
            p[e] *= inv;
            dout[OFF_LOG + token * NE + e]  = acc[e];
            dout[OFF_PROB + token * NE + e] = p[e];
        }
        // top-2 (ties -> lowest index, matching lax.top_k)
        int i1 = 0;
#pragma unroll
        for (int e = 1; e < NE; ++e) if (p[e] > p[i1]) i1 = e;
        int i2 = (i1 == 0) ? 1 : 0;
#pragma unroll
        for (int e = 0; e < NE; ++e) if (e != i1 && p[e] > p[i2]) i2 = e;
        float s2 = p[i1] + p[i2];
        top_e[token * 2 + 0] = i1;  top_w[token * 2 + 0] = p[i1] / s2;
        top_e[token * 2 + 1] = i2;  top_w[token * 2 + 1] = p[i2] / s2;
        atomicAdd(&counts[i1], 1);
        atomicAdd(&counts[i2], 1);
    }
}

// ---------------- kernel 3: deterministic parallel token-list build ---------
// One block, 256 threads = (expert e, segment s of 256 tokens).
// Pass 1: per-(e,s) counts into LDS. Pass 2: exact prefix -> fill in token
// order. Fully deterministic (no atomics), 256-way parallel.
#define NSEG 32
#define SEGSZ (TOK / NSEG)
__global__ void __launch_bounds__(256) k_build(
    int* __restrict__ offsets,
    const int* __restrict__ top_e, const float* __restrict__ top_w,
    int* __restrict__ ptok, float* __restrict__ pw) {
    __shared__ int segc[NE * NSEG];
    const int tid = threadIdx.x;
    const int e   = tid >> 5;        // 0..7
    const int seg = tid & 31;        // 0..31
    const int t0  = seg * SEGSZ;

    int c = 0;
    for (int t = t0; t < t0 + SEGSZ; ++t) {
#pragma unroll
        for (int k = 0; k < TOPK; ++k) c += (top_e[t * TOPK + k] == e);
    }
    segc[e * NSEG + seg] = c;
    __syncthreads();

    int base = 0;
    for (int j = 0; j < e * NSEG + seg; ++j) base += segc[j];
    if (seg == 0) offsets[e] = base;

    int pos = base;
    for (int t = t0; t < t0 + SEGSZ; ++t) {
#pragma unroll
        for (int k = 0; k < TOPK; ++k) {
            if (top_e[t * TOPK + k] == e) {
                ptok[pos] = t;
                pw[pos]   = top_w[t * TOPK + k];
                ++pos;
            }
        }
    }
}

// ---------------- kernel 4: routing statistics ------------------------------
__global__ void k_stats(const float* __restrict__ probs, const int* __restrict__ counts,
                        float* __restrict__ dout) {
    __shared__ float red[256];
    int tid = threadIdx.x;
    for (int e = 0; e < NE; ++e) {
        float s = 0.0f;
        for (int t = tid; t < TOK; t += 256) s += probs[t * NE + e];
        red[tid] = s;
        __syncthreads();
        for (int w = 128; w > 0; w >>= 1) {
            if (tid < w) red[tid] += red[tid + w];
            __syncthreads();
        }
        if (tid == 0) {
            dout[OFF_AP + e] = red[0] / (float)TOK;
            dout[OFF_EF + e] = (float)counts[e] / (float)(TOK * TOPK);
        }
        __syncthreads();
    }
}

// ---------------- kernel 5: fused SwiGLU + down projection ------------------
// grid: (256 tiles of 32 tokens, NE experts), 256 threads = 8 wave32.
// LDS: X[32][1032]bf16 + hidden[32][2056]bf16 ~= 193 KB of the 320 KB WGP LDS.
__global__ void __launch_bounds__(256) k_ffn(
    const float* __restrict__ x,
    const float* __restrict__ gw, const float* __restrict__ uw,
    const float* __restrict__ dw,
    const int* __restrict__ counts, const int* __restrict__ offsets,
    const int* __restrict__ ptok, const float* __restrict__ pw,
    float* __restrict__ out) {
    const int e    = blockIdx.y;
    const int tile = blockIdx.x;
    const int cnt  = counts[e];
    if (tile * MT >= cnt) return;     // over-launched tile: nothing to do

    extern __shared__ char smem[];
    __bf16* Xl   = (__bf16*)smem;                                       // [MT][XS]
    __bf16* Hl   = (__bf16*)(smem + (size_t)MT * XS * sizeof(__bf16));  // [MT][HS]
    float*  wrow = (float*)(smem + (size_t)MT * XS * sizeof(__bf16)
                                 + (size_t)MT * HS * sizeof(__bf16));   // [MT]
    int*    trow = (int*)(wrow + MT);                                   // [MT]

    const int tid  = threadIdx.x;
    const int lane = tid & 31;
    const int wv   = tid >> 5;
    const int half = lane >> 4;
    const int l16  = lane & 15;
    const int offe = offsets[e];

    if (tid < MT) {
        int pos = tile * MT + tid;
        int cl  = (pos < cnt) ? pos : (cnt - 1);        // clamp (dup row, weight 0)
        trow[tid] = ptok[offe + cl];
        wrow[tid] = (pos < cnt) ? pw[offe + pos] : 0.0f;
    }
    __syncthreads();

    // Stage X tile [MT, H] as bf16 into LDS (coalesced)
    for (int i = tid; i < MT * DH; i += blockDim.x) {
        int r = i >> 10;            // DH == 1024
        int h = i & (DH - 1);
        Xl[r * XS + h] = (__bf16)x[(size_t)trow[r] * DH + h];
    }
    __syncthreads();

    // ---- Phase 1: hidden[MT, I] = silu(X @ gateW^T) * (X @ upW^T) ----------
    // Each wave: 16 n-tiles; A fragments (2 m-tiles) shared by gate & up WMMAs.
    const float* gwe = gw + (size_t)e * DI * DH;
    const float* uwe = uw + (size_t)e * DI * DH;
    const __bf16* arowX0 = Xl + l16 * XS;
    const __bf16* arowX1 = Xl + (16 + l16) * XS;
    for (int nt = wv; nt < DI / 16; nt += 8) {
        const int ncol = nt * 16 + l16;
        const float* grow = gwe + (size_t)ncol * DH;
        const float* urow = uwe + (size_t)ncol * DH;
        if (nt + 8 < DI / 16) {         // prefetch next n-tile's weight rows
            __builtin_prefetch(grow + 128 * DH, 0, 1);
            __builtin_prefetch(urow + 128 * DH, 0, 1);
        }
        v8f cg0 = {}, cg1 = {}, cu0 = {}, cu1 = {};
        for (int ks = 0; ks < DH / 32; ++ks) {
            const int k0 = ks * 32;
            v16bf a0 = fragA_lds(arowX0, k0, half);
            v16bf a1 = fragA_lds(arowX1, k0, half);
            v16bf bg = fragB_g32(grow, k0, half);
            v16bf bu = fragB_g32(urow, k0, half);
            cg0 = wmma_bf16(a0, bg, cg0);
            cg1 = wmma_bf16(a1, bg, cg1);   // B reused across m-tiles
            cu0 = wmma_bf16(a0, bu, cu0);
            cu1 = wmma_bf16(a1, bu, cu1);
        }
#pragma unroll
        for (int r = 0; r < 8; ++r) {
            const int m0 = r + half * 8;
            Hl[m0 * HS + nt * 16 + l16]        = (__bf16)silu_mul(cg0[r], cu0[r]);
            Hl[(16 + m0) * HS + nt * 16 + l16] = (__bf16)silu_mul(cg1[r], cu1[r]);
        }
    }
    __syncthreads();

    // ---- Phase 2: out[MT, H] += combine * (hidden @ downW^T) ---------------
    // Wave owns H columns [wv*128, wv*128+128); 16 C fragments in registers.
    const float* dwe = dw + (size_t)e * DH * DI;
    v8f cacc[2][8];
#pragma unroll
    for (int mt = 0; mt < 2; ++mt)
#pragma unroll
        for (int j = 0; j < 8; ++j) cacc[mt][j] = (v8f){};
    const __bf16* arowH0 = Hl + l16 * HS;
    const __bf16* arowH1 = Hl + (16 + l16) * HS;
    for (int ks = 0; ks < DI / 32; ++ks) {
        const int k0 = ks * 32;
        v16bf a0 = fragA_lds(arowH0, k0, half);
        v16bf a1 = fragA_lds(arowH1, k0, half);
#pragma unroll
        for (int j = 0; j < 8; ++j) {
            const int hcol = wv * 128 + j * 16 + l16;
            v16bf b = fragB_g32(dwe + (size_t)hcol * DI, k0, half);
            cacc[0][j] = wmma_bf16(a0, b, cacc[0][j]);   // B reused across m-tiles
            cacc[1][j] = wmma_bf16(a1, b, cacc[1][j]);
        }
    }
    // Scale by combine weight; <=2 commutative atomics per output element
    // (one per selected expert) keeps the result bitwise deterministic.
#pragma unroll
    for (int mt = 0; mt < 2; ++mt) {
#pragma unroll
        for (int j = 0; j < 8; ++j) {
#pragma unroll
            for (int r = 0; r < 8; ++r) {
                const int m = mt * 16 + r + half * 8;
                const float v = cacc[mt][j][r] * wrow[m];
                atomicAdd(&out[(size_t)trow[m] * DH + (wv * 128 + j * 16 + l16)], v);
            }
        }
    }
}

// ---------------- host-side launcher ----------------------------------------
extern "C" void kernel_launch(void* const* d_in, const int* in_sizes, int n_in,
                              void* d_out, int out_size, void* d_ws, size_t ws_size,
                              hipStream_t stream) {
    const float* x   = (const float*)d_in[0];   // hidden_states [4,2048,1024]
    const float* rw  = (const float*)d_in[1];   // router_w [8,1024]
    const float* gw  = (const float*)d_in[2];   // gate_w [8,2048,1024]
    const float* uw  = (const float*)d_in[3];   // up_w   [8,2048,1024]
    const float* dwn = (const float*)d_in[4];   // down_w [8,1024,2048]
    float* out = (float*)d_out;

    char* ws = (char*)d_ws;
    int*   counts  = (int*)(ws + WS_COUNTS);
    int*   offsets = (int*)(ws + WS_OFFSETS);
    int*   tope    = (int*)(ws + WS_TOPE);
    float* topw    = (float*)(ws + WS_TOPW);
    int*   ptokl   = (int*)(ws + WS_PTOK);
    float* pwl     = (float*)(ws + WS_PW);

    k_zero<<<4096, 256, 0, stream>>>(out, counts);
    k_router<<<TOK / 8, 256, 0, stream>>>(x, rw, out, counts, tope, topw);
    k_build<<<1, 256, 0, stream>>>(offsets, tope, topw, ptokl, pwl);
    k_stats<<<1, 256, 0, stream>>>(out + OFF_PROB, counts, out);

    size_t shbytes = (size_t)MT * XS * sizeof(__bf16)
                   + (size_t)MT * HS * sizeof(__bf16)
                   + MT * sizeof(float) + MT * sizeof(int);   // ~193 KB / 320 KB WGP
    k_ffn<<<dim3(256, NE), 256, shbytes, stream>>>(x, gw, uw, dwn, counts, offsets,
                                                   ptokl, pwl, out);
}
